// SelfAttentionRPPEBlock_37890201485778
// MI455X (gfx1250) — compile-verified, matrix operand order checked by reading, not run on hardware
//
#include <hip/hip_runtime.h>
#include <hip/hip_bf16.h>

// ---------------------------------------------------------------------------
// CDNA5 (gfx1250) self-attention block, bf16 WMMA pipeline, double-buffered
// LDS with async-to-LDS copies and TDM tensor loads.
//   qkv = x @ Wqkv + bqkv            (f32 in -> bf16 out, f32 accum)
//   S   = softmax(Q K^T / sqrt(E))   (bf16 matmul, f32 softmax)
//   V'  = S @ V                      (bf16)
//   out = V' @ Wout + bout           (f32 out)
// ---------------------------------------------------------------------------

#define USE_ASYNC 1   // global_load_async_to_lds_b128 for direct bf16 tile copies
#define USE_TDM   1   // tensor_load_to_lds (TDM) for the score GEMM tiles

typedef __bf16 bf16_t;
typedef __attribute__((ext_vector_type(16))) __bf16 v16bf;
typedef __attribute__((ext_vector_type(8)))  __bf16 v8bf;
typedef __attribute__((ext_vector_type(4)))  __bf16 v4bf;
typedef __attribute__((ext_vector_type(8)))  float  v8f;
typedef __attribute__((ext_vector_type(4)))  unsigned int u32x4;
typedef __attribute__((ext_vector_type(8)))  int i32x8;
typedef __attribute__((ext_vector_type(4)))  int i32x4;

static constexpr int LDSS = 40;   // padded LDS row stride (bf16): 80B rows, conflict-free

// ---- fragment load: ISA 16-bit A/B 16x32 layout --------------------------
__device__ __forceinline__ v16bf load_frag(const bf16_t* base, int lane) {
  const int r  = lane & 15;
  const int kh = (lane >> 4) * 8;
  v8bf lo = *(const v8bf*)(base + r * LDSS + kh);
  v8bf hi = *(const v8bf*)(base + r * LDSS + 16 + kh);
  return __builtin_shufflevector(lo, hi, 0,1,2,3,4,5,6,7,8,9,10,11,12,13,14,15);
}

// ---- per-wave 32x64 WMMA subtile of a 128x128 block ----------------------
struct Acc { v8f c[2][4]; };

__device__ __forceinline__ void mma_tile(const bf16_t* As, const bf16_t* Bs,
                                         Acc& acc, int wave, int lane) {
  const int wm = (wave >> 1) * 32;
  const int wn = (wave & 1) * 64;
  v16bf a0 = load_frag(As + (wm +  0) * LDSS, lane);
  v16bf a1 = load_frag(As + (wm + 16) * LDSS, lane);
#pragma unroll
  for (int j = 0; j < 4; ++j) {
    v16bf b = load_frag(Bs + (wn + j * 16) * LDSS, lane);
    acc.c[0][j] = __builtin_amdgcn_wmma_f32_16x16x32_bf16(
        false, a0, false, b, (short)0, acc.c[0][j], false, false);
    acc.c[1][j] = __builtin_amdgcn_wmma_f32_16x16x32_bf16(
        false, a1, false, b, (short)0, acc.c[1][j], false, false);
  }
}

// ---- VGPR-staged loaders (for f32 -> bf16 converting tiles) --------------
// A tile: 128 rows x 32 K, f32 source
struct FA4 { float4 v[4]; };
__device__ __forceinline__ void fetchA_f32(FA4& f, const float* src, int ldg) {
  const int tid = threadIdx.x;
#pragma unroll
  for (int i = 0; i < 4; ++i) {
    int idx = tid + 256 * i;
    int row = idx >> 3;
    int k   = (idx & 7) * 4;
    f.v[i] = *(const float4*)(src + (size_t)row * ldg + k);
  }
}
__device__ __forceinline__ void commitA_f32(bf16_t* dst, const FA4& f) {
  const int tid = threadIdx.x;
#pragma unroll
  for (int i = 0; i < 4; ++i) {
    int idx = tid + 256 * i;
    int row = idx >> 3;
    int k   = (idx & 7) * 4;
    v4bf o;
    o[0] = (bf16_t)f.v[i].x; o[1] = (bf16_t)f.v[i].y;
    o[2] = (bf16_t)f.v[i].z; o[3] = (bf16_t)f.v[i].w;
    *(v4bf*)(dst + row * LDSS + k) = o;
  }
}
// B tile transposed: dst[n][k] = src[k*ldg + n], f32 source
struct FT16f { float v[16]; };
__device__ __forceinline__ void fetchT_f32(FT16f& f, const float* src, int ldg) {
  const int tid = threadIdx.x;
#pragma unroll
  for (int i = 0; i < 16; ++i) {
    int idx = tid + 256 * i;
    int n = idx & 127, k = idx >> 7;
    f.v[i] = src[(size_t)k * ldg + n];
  }
}
__device__ __forceinline__ void commitT_f32(bf16_t* dst, const FT16f& f) {
  const int tid = threadIdx.x;
#pragma unroll
  for (int i = 0; i < 16; ++i) {
    int idx = tid + 256 * i;
    int n = idx & 127, k = idx >> 7;
    dst[n * LDSS + k] = (bf16_t)f.v[i];
  }
}
// B tile transposed, bf16 source
struct FT16b { bf16_t v[16]; };
__device__ __forceinline__ void fetchT_bf16(FT16b& f, const bf16_t* src, int ldg) {
  const int tid = threadIdx.x;
#pragma unroll
  for (int i = 0; i < 16; ++i) {
    int idx = tid + 256 * i;
    int n = idx & 127, k = idx >> 7;
    f.v[i] = src[(size_t)k * ldg + n];
  }
}
__device__ __forceinline__ void commitT_bf16(bf16_t* dst, const FT16b& f) {
  const int tid = threadIdx.x;
#pragma unroll
  for (int i = 0; i < 16; ++i) {
    int idx = tid + 256 * i;
    int n = idx & 127, k = idx >> 7;
    dst[n * LDSS + k] = f.v[i];
  }
}

// ---- async global -> LDS copy of a 128x32 bf16 tile (ASYNCcnt) -----------
__device__ __forceinline__ void async_tile_bf16(bf16_t* dst, const bf16_t* src, int ldg) {
#if USE_ASYNC
  const int tid = threadIdx.x;
#pragma unroll
  for (int i = 0; i < 2; ++i) {
    int idx = tid + 256 * i;
    int row = idx >> 2;
    int k   = (idx & 3) * 8;
    unsigned lds = (unsigned)(uintptr_t)(dst + row * LDSS + k);  // LDS aperture: addr[31:0]
    const bf16_t* g = src + (size_t)row * ldg + k;
    asm volatile("global_load_async_to_lds_b128 %0, %1, off"
                 :: "v"(lds), "v"(g) : "memory");
  }
#else
  const int tid = threadIdx.x;
#pragma unroll
  for (int i = 0; i < 2; ++i) {
    int idx = tid + 256 * i;
    int row = idx >> 2;
    int k   = (idx & 3) * 8;
    *(v8bf*)(dst + row * LDSS + k) = *(const v8bf*)(src + (size_t)row * ldg + k);
  }
#endif
}
__device__ __forceinline__ void wait_async() {
#if USE_ASYNC
  asm volatile("s_wait_asynccnt 0x0" ::: "memory");
#endif
}

// ---- TDM: one descriptor moves a 128x32 bf16 tile into padded LDS --------
// pad_enable with interval=64B (one tile row), amount=16B -> LDS stride 40 elems.
__device__ __forceinline__ void tdm_tile_bf16(const bf16_t* gsrc, bf16_t* ldst, int ldg) {
  unsigned long long ga = (unsigned long long)(uintptr_t)gsrc;
  u32x4 g0;
  g0[0] = 1u;                                            // count=1, user mode
  g0[1] = (unsigned)(uintptr_t)ldst;                     // lds_addr (aperture low bits)
  g0[2] = (unsigned)(ga & 0xffffffffu);                  // global_addr[31:0]
  g0[3] = (unsigned)((ga >> 32) & 0x01ffffffu) | 0x80000000u;  // addr[56:32] | type=2
  i32x8 g1;
  g1[0] = 0x06D10000;         // data_size=2B, pad_enable, pad_interval=16DW, pad_amount=4DW
  g1[1] = (32  << 16);        // tensor_dim0 = 32 (low16 in [63:48])
  g1[2] = (128 << 16);        // tensor_dim0 hi=0, tensor_dim1 = 128 (low16 in [95:80])
  g1[3] = (32  << 16);        // tensor_dim1 hi=0, tile_dim0 = 32
  g1[4] = 128;                // tile_dim1 = 128, tile_dim2 = 0
  g1[5] = ldg;                // tensor_dim0_stride[31:0] (elements)
  g1[6] = 0;                  // stride hi, tensor_dim1_stride lo
  g1[7] = 0;
  i32x4 z4; z4[0] = 0; z4[1] = 0; z4[2] = 0; z4[3] = 0;  // groups 2/3 unused (2D)
  i32x8 z8; z8[0] = 0; z8[1] = 0; z8[2] = 0; z8[3] = 0;
            z8[4] = 0; z8[5] = 0; z8[6] = 0; z8[7] = 0;
  __builtin_amdgcn_tensor_load_to_lds(g0, g1, z4, z4, z8, 0);
}

// =========================== Kernel 1: QKV GEMM ===========================
// C[16384,3072] = x[16384,1024] @ Wqkv[1024,3072] + bqkv  -> bf16
__global__ __launch_bounds__(256) void k_gemm_qkv(const float* __restrict__ x,
                                                  const float* __restrict__ W,
                                                  const float* __restrict__ bias,
                                                  bf16_t* __restrict__ out) {
  __shared__ alignas(16) bf16_t As[2][128 * LDSS];
  __shared__ alignas(16) bf16_t Bs[2][128 * LDSS];
  const int nTile = blockIdx.x * 128;
  const int mTile = blockIdx.y * 128;
  const int tid = threadIdx.x, wave = tid >> 5, lane = tid & 31;
  const float* aSrc = x + (size_t)mTile * 1024;
  const float* bSrc = W + nTile;
  Acc acc = {};
  FA4 fa; FT16f fb;
  fetchA_f32(fa, aSrc, 1024);
  fetchT_f32(fb, bSrc, 3072);
  commitA_f32(As[0], fa);
  commitT_f32(Bs[0], fb);
  __syncthreads();
  int cur = 0;
  for (int kt = 0; kt < 32; ++kt) {
    const bool more = (kt + 1) < 32;
    if (more) {
      fetchA_f32(fa, aSrc + (kt + 1) * 32, 1024);
      fetchT_f32(fb, bSrc + (size_t)(kt + 1) * 32 * 3072, 3072);
    }
    mma_tile(As[cur], Bs[cur], acc, wave, lane);
    if (more) {
      commitA_f32(As[cur ^ 1], fa);
      commitT_f32(Bs[cur ^ 1], fb);
      __syncthreads();
      cur ^= 1;
    }
  }
  const int wm = (wave >> 1) * 32, wn = (wave & 1) * 64;
  const int rOff = (lane >> 4) << 3, col0 = lane & 15;
#pragma unroll
  for (int i = 0; i < 2; ++i)
#pragma unroll
    for (int j = 0; j < 4; ++j)
#pragma unroll
      for (int r = 0; r < 8; ++r) {
        int row = mTile + wm + i * 16 + rOff + r;
        int col = nTile + wn + j * 16 + col0;
        out[(size_t)row * 3072 + col] = (bf16_t)(acc.c[i][j][r] + bias[col]);
      }
}

// =========================== Kernel 2: Q K^T (TDM) ========================
__global__ __launch_bounds__(256) void k_gemm_scores(const bf16_t* __restrict__ qkv,
                                                     bf16_t* __restrict__ S) {
  __shared__ alignas(16) bf16_t As[2][128 * LDSS];
  __shared__ alignas(16) bf16_t Bs[2][128 * LDSS];
  const int tTile = blockIdx.x * 128;
  const int qTile = blockIdx.y * 128;
  const int b = blockIdx.z;
  const bf16_t* qp = qkv + (size_t)b * 2048 * 3072 + (size_t)qTile * 3072;        // q rows
  const bf16_t* kp = qkv + (size_t)b * 2048 * 3072 + (size_t)tTile * 3072 + 1024; // k rows
  const int tid = threadIdx.x, wave = tid >> 5, lane = tid & 31;
  Acc acc = {};
#if USE_TDM
  if (wave == 0) {
    tdm_tile_bf16(qp, As[0], 3072);
    tdm_tile_bf16(kp, Bs[0], 3072);
    __builtin_amdgcn_s_wait_tensorcnt(0);
  }
  __syncthreads();
  int cur = 0;
  for (int kt = 0; kt < 32; ++kt) {
    const bool more = (kt + 1) < 32;
    if (more && wave == 0) {
      tdm_tile_bf16(qp + (kt + 1) * 32, As[cur ^ 1], 3072);
      tdm_tile_bf16(kp + (kt + 1) * 32, Bs[cur ^ 1], 3072);
    }
    mma_tile(As[cur], Bs[cur], acc, wave, lane);
    if (more) {
      if (wave == 0) __builtin_amdgcn_s_wait_tensorcnt(0);
      __syncthreads();
      cur ^= 1;
    }
  }
#else
  async_tile_bf16(As[0], qp, 3072);
  async_tile_bf16(Bs[0], kp, 3072);
  wait_async();
  __syncthreads();
  int cur = 0;
  for (int kt = 0; kt < 32; ++kt) {
    const bool more = (kt + 1) < 32;
    if (more) {
      async_tile_bf16(As[cur ^ 1], qp + (kt + 1) * 32, 3072);
      async_tile_bf16(Bs[cur ^ 1], kp + (kt + 1) * 32, 3072);
    }
    mma_tile(As[cur], Bs[cur], acc, wave, lane);
    if (more) { wait_async(); __syncthreads(); cur ^= 1; }
  }
#endif
  const float scale = 0.03125f;  // 1/sqrt(1024); +B logit bias is a softmax no-op
  bf16_t* Sp = S + (size_t)b * 2048 * 2048;
  const int wm = (wave >> 1) * 32, wn = (wave & 1) * 64;
  const int rOff = (lane >> 4) << 3, col0 = lane & 15;
#pragma unroll
  for (int i = 0; i < 2; ++i)
#pragma unroll
    for (int j = 0; j < 4; ++j)
#pragma unroll
      for (int r = 0; r < 8; ++r) {
        int row = qTile + wm + i * 16 + rOff + r;
        int col = tTile + wn + j * 16 + col0;
        Sp[(size_t)row * 2048 + col] = (bf16_t)(acc.c[i][j][r] * scale);
      }
}

// =========================== Kernel 3: softmax ============================
__global__ __launch_bounds__(256) void k_softmax(bf16_t* __restrict__ S) {
  bf16_t* p = S + (size_t)blockIdx.x * 2048;
  const int tid = threadIdx.x, lane = tid & 31, wave = tid >> 5;
  __shared__ float redm[8];
  __shared__ float reds[8];
  float v[8];
  float m = -3.4e38f;
#pragma unroll
  for (int i = 0; i < 8; ++i) { v[i] = (float)p[tid + 256 * i]; m = fmaxf(m, v[i]); }
#pragma unroll
  for (int off = 16; off; off >>= 1) m = fmaxf(m, __shfl_xor(m, off, 32));
  if (lane == 0) redm[wave] = m;
  __syncthreads();
  if (wave == 0) {
    float t = redm[lane & 7];
#pragma unroll
    for (int off = 4; off; off >>= 1) t = fmaxf(t, __shfl_xor(t, off, 32));
    if (lane == 0) redm[0] = t;
  }
  __syncthreads();
  m = redm[0];
  float s = 0.f;
#pragma unroll
  for (int i = 0; i < 8; ++i) { v[i] = __expf(v[i] - m); s += v[i]; }
#pragma unroll
  for (int off = 16; off; off >>= 1) s += __shfl_xor(s, off, 32);
  if (lane == 0) reds[wave] = s;
  __syncthreads();
  if (wave == 0) {
    float t = reds[lane & 7];
#pragma unroll
    for (int off = 4; off; off >>= 1) t += __shfl_xor(t, off, 32);
    if (lane == 0) reds[0] = t;
  }
  __syncthreads();
  const float inv = 1.0f / reds[0];
#pragma unroll
  for (int i = 0; i < 8; ++i) p[tid + 256 * i] = (bf16_t)(v[i] * inv);
}

// =========================== Kernel 4: P @ V ==============================
__global__ __launch_bounds__(256) void k_gemm_values(const bf16_t* __restrict__ S,
                                                     const bf16_t* __restrict__ qkv,
                                                     bf16_t* __restrict__ Vout) {
  __shared__ alignas(16) bf16_t As[2][128 * LDSS];
  __shared__ alignas(16) bf16_t Bs[2][128 * LDSS];
  const int eTile = blockIdx.x * 128;
  const int qTile = blockIdx.y * 128;
  const int b = blockIdx.z;
  const bf16_t* P  = S   + (size_t)b * 2048 * 2048 + (size_t)qTile * 2048;
  const bf16_t* vp = qkv + (size_t)b * 2048 * 3072 + 2048 + eTile;   // v slice
  const int tid = threadIdx.x, wave = tid >> 5, lane = tid & 31;
  Acc acc = {};
  FT16b fb;
  async_tile_bf16(As[0], P, 2048);
  fetchT_bf16(fb, vp, 3072);
  commitT_bf16(Bs[0], fb);
  wait_async();
  __syncthreads();
  int cur = 0;
  for (int kt = 0; kt < 64; ++kt) {                  // 2048 / 32 key steps
    const bool more = (kt + 1) < 64;
    if (more) {
      async_tile_bf16(As[cur ^ 1], P + (kt + 1) * 32, 2048);
      fetchT_bf16(fb, vp + (size_t)(kt + 1) * 32 * 3072, 3072);
    }
    mma_tile(As[cur], Bs[cur], acc, wave, lane);
    if (more) {
      commitT_bf16(Bs[cur ^ 1], fb);
      wait_async();
      __syncthreads();
      cur ^= 1;
    }
  }
  const int wm = (wave >> 1) * 32, wn = (wave & 1) * 64;
  const int rOff = (lane >> 4) << 3, col0 = lane & 15;
#pragma unroll
  for (int i = 0; i < 2; ++i)
#pragma unroll
    for (int j = 0; j < 4; ++j)
#pragma unroll
      for (int r = 0; r < 8; ++r) {
        int row = qTile + wm + i * 16 + rOff + r;
        int col = eTile + wn + j * 16 + col0;
        Vout[((size_t)b * 2048 + row) * 1024 + col] = (bf16_t)acc.c[i][j][r];
      }
}

// =========================== Kernel 5: out proj ===========================
__global__ __launch_bounds__(256) void k_gemm_out(const bf16_t* __restrict__ V,
                                                  const float* __restrict__ W,
                                                  const float* __restrict__ bias,
                                                  float* __restrict__ out) {
  __shared__ alignas(16) bf16_t As[2][128 * LDSS];
  __shared__ alignas(16) bf16_t Bs[2][128 * LDSS];
  const int nTile = blockIdx.x * 128;
  const int mTile = blockIdx.y * 128;
  const int tid = threadIdx.x, wave = tid >> 5, lane = tid & 31;
  const bf16_t* aSrc = V + (size_t)mTile * 1024;
  const float*  bSrc = W + nTile;
  Acc acc = {};
  FT16f fb;
  async_tile_bf16(As[0], aSrc, 1024);
  fetchT_f32(fb, bSrc, 1024);
  commitT_f32(Bs[0], fb);
  wait_async();
  __syncthreads();
  int cur = 0;
  for (int kt = 0; kt < 32; ++kt) {
    const bool more = (kt + 1) < 32;
    if (more) {
      async_tile_bf16(As[cur ^ 1], aSrc + (kt + 1) * 32, 1024);
      fetchT_f32(fb, bSrc + (size_t)(kt + 1) * 32 * 1024, 1024);
    }
    mma_tile(As[cur], Bs[cur], acc, wave, lane);
    if (more) {
      commitT_f32(Bs[cur ^ 1], fb);
      wait_async();
      __syncthreads();
      cur ^= 1;
    }
  }
  const int wm = (wave >> 1) * 32, wn = (wave & 1) * 64;
  const int rOff = (lane >> 4) << 3, col0 = lane & 15;
#pragma unroll
  for (int i = 0; i < 2; ++i)
#pragma unroll
    for (int j = 0; j < 4; ++j)
#pragma unroll
      for (int r = 0; r < 8; ++r) {
        int row = mTile + wm + i * 16 + rOff + r;
        int col = nTile + wn + j * 16 + col0;
        out[(size_t)row * 1024 + col] = acc.c[i][j][r] + bias[col];
      }
}

// =========================== launch =======================================
extern "C" void kernel_launch(void* const* d_in, const int* in_sizes, int n_in,
                              void* d_out, int out_size, void* d_ws, size_t ws_size,
                              hipStream_t stream) {
  const float* x    = (const float*)d_in[0];
  // d_in[1] ("b") is shadowed/unused in the reference forward.
  const float* Wqkv = (const float*)d_in[2];
  const float* bqkv = (const float*)d_in[3];
  const float* Wout = (const float*)d_in[4];
  const float* bout = (const float*)d_in[5];
  float* out = (float*)d_out;

  // workspace layout (bf16): qkv [16384,3072] | S [8,2048,2048] | V' [16384,1024]
  bf16_t* qkv = (bf16_t*)d_ws;
  bf16_t* S   = qkv + (size_t)16384 * 3072;
  bf16_t* Vp  = S   + (size_t)8 * 2048 * 2048;

  dim3 blk(256);
  k_gemm_qkv   <<<dim3(24, 128),   blk, 0, stream>>>(x, Wqkv, bqkv, qkv);
  k_gemm_scores<<<dim3(16, 16, 8), blk, 0, stream>>>(qkv, S);
  k_softmax    <<<dim3(16384),     blk, 0, stream>>>(S);
  k_gemm_values<<<dim3(8, 16, 8),  blk, 0, stream>>>(S, qkv, Vp);
  k_gemm_out   <<<dim3(8, 128),    blk, 0, stream>>>(Vp, Wout, bout, out);
}
